// VectorQuantizer_67791763800755
// MI455X (gfx1250) — matrix operand order repaired and checked by reading, not compiled, not run on hardware
//
#include <hip/hip_runtime.h>

#define N_ROWS 16384
#define DIM 256
#define K_CODES 8192
#define ROWS_PER_WG 128
#define CODE_BLOCK 1024
#define CODE_TILE 64
#define BETA 0.25f
#define EPS_V 1e-5f
#define EOP_ID 3
#define PAD_ID 2

typedef unsigned short u16;
typedef unsigned int u32;
typedef unsigned long long u64;

typedef __attribute__((ext_vector_type(16))) __bf16 v16bf;
typedef __attribute__((ext_vector_type(8)))  float  v8f;
typedef __attribute__((ext_vector_type(4)))  u32    v4u;
typedef __attribute__((ext_vector_type(8)))  int    v8i;
typedef __attribute__((ext_vector_type(4)))  int    v4i;

#if defined(__has_builtin)
#if __has_builtin(__builtin_amdgcn_tensor_load_to_lds) && \
    __has_builtin(__builtin_amdgcn_s_wait_tensorcnt)
#define HAVE_TDM 1
#endif
#endif
#ifndef HAVE_TDM
#define HAVE_TDM 0
#endif

union frag16 { uint4 q[2]; v16bf v; };

__device__ __forceinline__ u16 f32_to_bf16_rne(float f) {
  u32 u = __float_as_uint(f);
  u32 r = u + 0x7FFFu + ((u >> 16) & 1u);
  return (u16)(r >> 16);
}
__device__ __forceinline__ float bf16_to_f32(u16 h) {
  return __uint_as_float(((u32)h) << 16);
}

#if HAVE_TDM
// Issue a TDM DMA of `nelem` contiguous bf16 elements global -> LDS.
// D# group0: count=1, lds_addr, global_addr, type=2 ("image").
// D# group1: data_size=2B, 1-D tile (tile_dim0=nelem, tile_dim1=1),
//            tensor_dim0 large enough that no OOB clamping occurs.
__device__ __forceinline__ void tdm_load_tile(const u16* gsrc, u16* lds_dst,
                                              u32 nelem) {
  u64 ga = (u64)(uintptr_t)gsrc;
  u32 lds = (u32)(uintptr_t)lds_dst;  // low 32 bits of flat ptr == LDS offset
  v4u g0 = {1u,                                   // count=1, user mode
            lds,                                  // lds_addr (bytes)
            (u32)ga,                              // global_addr[31:0]
            ((u32)(ga >> 32) & 0x01FFFFFFu) | 0x80000000u};  // [56:32], type=2
  const u32 dim0 = (u32)K_CODES * (u32)DIM;       // whole plane, in 2B units
  v8i g1 = {(int)0x00010000u,                     // wg_mask=0, data_size=1(2B)
            (int)((dim0 & 0xFFFFu) << 16),        // tensor_dim0[15:0]
            (int)((dim0 >> 16) | (1u << 16)),     // tensor_dim0[31:16], dim1=1
            (int)(nelem << 16),                   // tile_dim0
            1,                                    // tile_dim1=1, tile_dim2=0
            (int)dim0,                            // tensor_dim0_stride[31:0]
            0, 0};
  v4i gz = {0, 0, 0, 0};
#if __clang_major__ >= 23
  v8i gz8 = {0, 0, 0, 0, 0, 0, 0, 0};
  __builtin_amdgcn_tensor_load_to_lds(g0, g1, gz, gz, gz8, 0);
#else
  __builtin_amdgcn_tensor_load_to_lds(g0, g1, gz, gz, 0);
#endif
}
#endif

// ---------------- prep kernels ----------------

__global__ void init_best_kernel(u64* best) {
  int i = blockIdx.x * blockDim.x + threadIdx.x;
  if (i < N_ROWS) best[i] = 0xFFFFFFFFFFFFFFFFull;
}

__global__ void split_kernel(const float* __restrict__ src,
                             u16* __restrict__ hi, u16* __restrict__ lo, int n) {
  int i = blockIdx.x * blockDim.x + threadIdx.x;
  if (i < n) {
    float v = src[i];
    u16 h = f32_to_bf16_rne(v);
    float r = v - bf16_to_f32(h);
    hi[i] = h;
    lo[i] = f32_to_bf16_rne(r);
  }
}

__global__ void e2_kernel(const float* __restrict__ cb, float* __restrict__ e2) {
  __shared__ float red[64];
  int k = blockIdx.x;
  int t = threadIdx.x;
  const float* r = cb + (size_t)k * DIM;
  float s = 0.0f;
  for (int d = t; d < DIM; d += 64) { float v = r[d]; s += v * v; }
  red[t] = s;
  __syncthreads();
  for (int w = 32; w > 0; w >>= 1) {
    if (t < w) red[t] += red[t + w];
    __syncthreads();
  }
  if (t == 0) e2[k] = red[0];
}

// ---------------- main fused GEMM + argmin ----------------

__global__ __launch_bounds__(256) void vq_argmin_kernel(
    const u16* __restrict__ z_hi, const u16* __restrict__ z_lo,
    const u16* __restrict__ cb_hi, const u16* __restrict__ cb_lo,
    const float* __restrict__ e2, u64* __restrict__ best) {
  __shared__ u16 s_cbhi[CODE_TILE * DIM];  // 32 KB
  __shared__ u16 s_cblo[CODE_TILE * DIM];  // 32 KB

  const int tid  = threadIdx.x;
  const int wave = tid >> 5;
  const int lane = tid & 31;
  const int l16  = lane & 15;
  const int hh   = lane >> 4;  // half-wave selector

  const int rowBase  = blockIdx.x * ROWS_PER_WG + wave * 16;
  const int row      = rowBase + l16;
  const int codeBase = blockIdx.y * CODE_BLOCK;

  // A fragments (z rows, hi+lo) resident in VGPRs for the whole kernel.
  // 16-bit A layout: lane(l16)=M; VGPR0-3 hold K=8h..8h+7, VGPR4-7 hold K=16+8h..
  v16bf a_hi[8], a_lo[8];
  {
    const u16* zr_hi = z_hi + (size_t)row * DIM;
    const u16* zr_lo = z_lo + (size_t)row * DIM;
#pragma unroll
    for (int dc = 0; dc < 8; ++dc) {
      int off0 = dc * 32 + 8 * hh;
      frag16 t;
      t.q[0] = *(const uint4*)(zr_hi + off0);
      t.q[1] = *(const uint4*)(zr_hi + off0 + 16);
      a_hi[dc] = t.v;
      t.q[0] = *(const uint4*)(zr_lo + off0);
      t.q[1] = *(const uint4*)(zr_lo + off0 + 16);
      a_lo[dc] = t.v;
    }
  }

  float best_d[8];
  u32   best_c[8];
#pragma unroll
  for (int j = 0; j < 8; ++j) { best_d[j] = 3.4e38f; best_c[j] = 0u; }

  const int nTiles = CODE_BLOCK / CODE_TILE;
  for (int t = 0; t < nTiles; ++t) {
    const int tileBase = codeBase + t * CODE_TILE;
    __syncthreads();  // prior tile fully consumed before overwrite

#if HAVE_TDM
    // Tensor Data Mover: DMA both 32KB bf16 planes straight into LDS.
    if (wave == 0) {
      tdm_load_tile(cb_hi + (size_t)tileBase * DIM, s_cbhi, CODE_TILE * DIM);
      tdm_load_tile(cb_lo + (size_t)tileBase * DIM, s_cblo, CODE_TILE * DIM);
      __builtin_amdgcn_s_wait_tensorcnt(0);
    }
#else
    {
      const uint4* ghi = (const uint4*)(cb_hi + (size_t)tileBase * DIM);
      const uint4* glo = (const uint4*)(cb_lo + (size_t)tileBase * DIM);
      uint4* shi = (uint4*)s_cbhi;
      uint4* slo = (uint4*)s_cblo;
      for (int i = tid; i < CODE_TILE * DIM / 8; i += 256) {
        shi[i] = ghi[i];
        slo[i] = glo[i];
      }
    }
#endif
    __syncthreads();

    // Prefetch next tile toward L2 (global_prefetch_b8).
    if (t + 1 < nTiles) {
      const u16* nh = cb_hi + (size_t)(tileBase + CODE_TILE) * DIM;
      const u16* nl = cb_lo + (size_t)(tileBase + CODE_TILE) * DIM;
      __builtin_prefetch(nh + (size_t)tid * 32, 0, 1);
      __builtin_prefetch(nl + (size_t)tid * 32, 0, 1);
    }

#pragma unroll
    for (int ct = 0; ct < CODE_TILE / 16; ++ct) {
      v8f acc = {};
      // 16-bit B layout: lane(l16)=N(code); VGPR0-3 hold K=8h..8h+7, VGPR4-7 hold K=16+8h..
      const u16* brow_hi = s_cbhi + (size_t)(ct * 16 + l16) * DIM;
      const u16* brow_lo = s_cblo + (size_t)(ct * 16 + l16) * DIM;
#pragma unroll
      for (int dc = 0; dc < 8; ++dc) {
        int offB = dc * 32 + 8 * hh;
        frag16 bh, bl;
        bh.q[0] = *(const uint4*)(brow_hi + offB);
        bh.q[1] = *(const uint4*)(brow_hi + offB + 16);
        bl.q[0] = *(const uint4*)(brow_lo + offB);
        bl.q[1] = *(const uint4*)(brow_lo + offB + 16);
        // 3xBF16 split: hi*hi + hi*lo + lo*hi, f32 accumulate.
        acc = __builtin_amdgcn_wmma_f32_16x16x32_bf16(false, a_hi[dc], false, bh.v,
                                                      (short)0, acc, false, false);
        acc = __builtin_amdgcn_wmma_f32_16x16x32_bf16(false, a_hi[dc], false, bl.v,
                                                      (short)0, acc, false, false);
        acc = __builtin_amdgcn_wmma_f32_16x16x32_bf16(false, a_lo[dc], false, bh.v,
                                                      (short)0, acc, false, false);
      }
      // C layout: lane = column (code), VGPR j = row j (lanes 0-15) / row 8+j.
      const int code = tileBase + ct * 16 + l16;
      const float e2v = e2[code];
#pragma unroll
      for (int j = 0; j < 8; ++j) {
        float dist = e2v - 2.0f * acc[j];
        if (dist < best_d[j]) { best_d[j] = dist; best_c[j] = (u32)code; }
      }
    }
  }

  // Cross-lane argmin over the 16 column slots, then global 64-bit atomic min.
#pragma unroll
  for (int j = 0; j < 8; ++j) {
    u32 ud = __float_as_uint(best_d[j]);
    ud = (ud & 0x80000000u) ? ~ud : (ud | 0x80000000u);  // sortable float
    u64 key = ((u64)ud << 32) | (u64)best_c[j];
    for (int m = 1; m < 16; m <<= 1) {
      u32 klo = (u32)key, khi = (u32)(key >> 32);
      u32 olo = (u32)__shfl_xor((int)klo, m, 32);
      u32 ohi = (u32)__shfl_xor((int)khi, m, 32);
      u64 other = ((u64)ohi << 32) | (u64)olo;
      if (other < key) key = other;
    }
    if (l16 == 0) {
      int r = rowBase + hh * 8 + j;
      atomicMin(best + r, key);
    }
  }
}

// ---------------- finalize: gather z_q, per-row loss, indices ----------------

__global__ __launch_bounds__(256) void finalize_kernel(
    const float* __restrict__ z, const float* __restrict__ cb,
    const u64* __restrict__ best, float* __restrict__ out_zq,
    float* __restrict__ out_idx, float* __restrict__ row_loss) {
  __shared__ float red[256];
  const int n = blockIdx.x;
  const int t = threadIdx.x;
  const u32 k = (u32)(best[n] & 0xFFFFFFFFull);
  const float q  = cb[(size_t)k * DIM + t];
  const float zv = z[(size_t)n * DIM + t];
  out_zq[(size_t)n * DIM + t] = q;  // z_q_ste forward value == z_q
  const float d = q - zv;
  red[t] = d * d;
  __syncthreads();
  for (int w = 128; w > 0; w >>= 1) {
    if (t < w) red[t] += red[t + w];
    __syncthreads();
  }
  if (t == 0) {
    row_loss[n] = red[0];
    out_idx[n] = (float)k;
  }
}

// ---------------- scalars: vq_loss + perplexity (deterministic) ----------------

__global__ __launch_bounds__(256) void scalars_kernel(
    const float* __restrict__ ema, const float* __restrict__ row_loss,
    float* __restrict__ out_loss, float* __restrict__ out_perp) {
  __shared__ float sS[256], sL[256], sQ[256];
  const int t = threadIdx.x;
  float S = 0.0f, L = 0.0f;
  for (int k = t; k < K_CODES; k += 256) {
    float c = ema[k];
    if (k == EOP_ID || k == PAD_ID) c = 0.0f;
    c = fmaxf(c, EPS_V);
    S += c;
    L += c * logf(c);
  }
  float Q = 0.0f;
  for (int n = t; n < N_ROWS; n += 256) Q += row_loss[n];
  sS[t] = S; sL[t] = L; sQ[t] = Q;
  __syncthreads();
  for (int w = 128; w > 0; w >>= 1) {
    if (t < w) { sS[t] += sS[t + w]; sL[t] += sL[t + w]; sQ[t] += sQ[t + w]; }
    __syncthreads();
  }
  if (t == 0) {
    float T = sS[0] + EPS_V;
    float entropy = (sS[0] * logf(T) - sL[0]) / T;
    *out_perp = expf(entropy);
    *out_loss = (1.0f + BETA) * sQ[0] / (float)(N_ROWS * DIM);
  }
}

// ---------------- launcher ----------------

extern "C" void kernel_launch(void* const* d_in, const int* in_sizes, int n_in,
                              void* d_out, int out_size, void* d_ws, size_t ws_size,
                              hipStream_t stream) {
  const float* z   = (const float*)d_in[0];   // (4,4096,256)
  const float* cb  = (const float*)d_in[1];   // (8192,256)
  const float* ema = (const float*)d_in[2];   // (8192,)

  char* ws = (char*)d_ws;
  u16* z_hi  = (u16*)(ws + 0);           // 8 MB
  u16* z_lo  = (u16*)(ws + 8388608);     // 8 MB
  u16* cb_hi = (u16*)(ws + 16777216);    // 4 MB
  u16* cb_lo = (u16*)(ws + 20971520);    // 4 MB
  float* e2  = (float*)(ws + 25165824);  // 32 KB
  u64* best  = (u64*)(ws + 25198592);    // 128 KB
  float* row_loss = (float*)(ws + 25329664);  // 64 KB

  float* out_zq   = (float*)d_out;            // 4194304 floats
  float* out_loss = out_zq + (size_t)N_ROWS * DIM;
  float* out_idx  = out_loss + 1;             // 16384 floats
  float* out_perp = out_idx + N_ROWS;

  init_best_kernel<<<(N_ROWS + 255) / 256, 256, 0, stream>>>(best);
  split_kernel<<<(N_ROWS * DIM) / 256, 256, 0, stream>>>(z, z_hi, z_lo, N_ROWS * DIM);
  split_kernel<<<(K_CODES * DIM) / 256, 256, 0, stream>>>(cb, cb_hi, cb_lo, K_CODES * DIM);
  e2_kernel<<<K_CODES, 64, 0, stream>>>(cb, e2);

  dim3 grid(N_ROWS / ROWS_PER_WG, K_CODES / CODE_BLOCK);  // 128 x 8
  vq_argmin_kernel<<<grid, 256, 0, stream>>>(z_hi, z_lo, cb_hi, cb_lo, e2, best);

  finalize_kernel<<<N_ROWS, 256, 0, stream>>>(z, cb, best, out_zq, out_idx, row_loss);
  scalars_kernel<<<1, 256, 0, stream>>>(ema, row_loss, out_loss, out_perp);
}
